// AttentionBlock_54030688584320
// MI455X (gfx1250) — matrix operand order repaired and checked by reading, not compiled, not run on hardware
//
#include <hip/hip_runtime.h>
#include <hip/hip_bf16.h>

typedef __attribute__((ext_vector_type(16))) _Float16 v16h;
typedef __attribute__((ext_vector_type(8)))  float    v8f;
typedef __attribute__((ext_vector_type(4)))  float    v4f;
typedef __attribute__((ext_vector_type(4)))  _Float16 v4h;

constexpr int B_  = 4;
constexpr int S_  = 2048;
constexpr int H_  = 1024;
constexpr int NH_ = 16;
constexpr int HD_ = 64;   // head dim

// Async 16-byte global -> LDS copy (CDNA5, tracked by ASYNCcnt).
__device__ __forceinline__ void async_b128(const _Float16* g, _Float16* lds_ptr) {
    uint32_t loff = (uint32_t)(uintptr_t)lds_ptr;   // flat LDS addr low bits == LDS offset
    asm volatile("global_load_async_to_lds_b128 %0, %1, off"
                 :: "v"(loff), "v"(g)
                 : "memory");
}
__device__ __forceinline__ void wait_async0() {
    asm volatile("s_wait_asynccnt 0x0" ::: "memory");
}

// ---------------------------------------------------------------------------
// Kernel 1: QKV projection.  OUT(f16) = A(f32,[B*S,H]) * W + b
//   vt_layout == 0 : out is [B, NH, S, HD]   (Q, K)
//   vt_layout == 1 : out is [B, NH, HD, S]   (V, pre-transposed per head)
// Block = 128 threads (4 waves).  128x64 output tile; each wave owns 32 rows
// (2 strips) x 64 cols = 8 WMMA accumulators.  K-step 32.
// LDS: A row-major [row][k]; W N-major [n][k] so the WMMA B-fragment
// (b[e] = B[k=half*16+e][n=c], this code's k order) is one contiguous 32B read.
// ---------------------------------------------------------------------------
__global__ __launch_bounds__(128)
void qkv_gemm_kernel(const float* __restrict__ A, const float* __restrict__ W,
                     const float* __restrict__ bias, _Float16* __restrict__ out,
                     float out_scale, int vt_layout)
{
    __shared__ _Float16 As[128][32];  // [row][k]   8 KB
    __shared__ _Float16 Bs[64][32];   // [n][k]     4 KB

    const int tid  = threadIdx.x;
    const int lane = tid & 31;
    const int wv   = tid >> 5;
    const int half = lane >> 4;
    const int c    = lane & 15;

    const int row0 = blockIdx.x * 128;  // [0, B*S)
    const int col0 = blockIdx.y * 64;   // [0, H); one head per 64-col strip

    // Per-thread staging sources (advance by 32 floats per K-step).
    const float* asrc = &A[(size_t)(row0 + tid) * H_];            // one row each
    const int    wkk  = tid >> 2;            // 0..31
    const int    wn0  = (tid & 3) * 16;      // 0,16,32,48
    const float* wsrc0 = &W[(size_t)wkk * H_ + col0 + wn0];

    v8f acc[2][4] = {};

    for (int k0 = 0; k0 < H_; k0 += 32) {
        __syncthreads();
        // --- stage A row: 8 x float4 -> 8 x b64 f16 stores (fully unrolled)
        const float* ap = asrc + k0;
#pragma unroll
        for (int u = 0; u < 8; ++u) {
            v4f x = *(const v4f*)(ap + u * 4);
            v4h y;
            y[0] = (_Float16)x[0]; y[1] = (_Float16)x[1];
            y[2] = (_Float16)x[2]; y[3] = (_Float16)x[3];
            *(v4h*)&As[tid][u * 4] = y;
        }
        // --- stage W: coalesced float4 row loads, N-major scatter to LDS
        const float* wp = wsrc0 + (size_t)k0 * H_;
#pragma unroll
        for (int u = 0; u < 4; ++u) {
            v4f x = *(const v4f*)(wp + u * 4);
#pragma unroll
            for (int e2 = 0; e2 < 4; ++e2)
                Bs[wn0 + u * 4 + e2][wkk] = (_Float16)x[e2];
        }
        if (k0 + 32 < H_) {                 // global_prefetch_b8 hints
            __builtin_prefetch(ap + 32, 0, 1);
            __builtin_prefetch(wp + 32 * H_, 0, 1);
        }
        __syncthreads();

        // A fragments: row = c; k = half*16+e
        v16h a[2];
#pragma unroll
        for (int st = 0; st < 2; ++st) {
            const int r = wv * 32 + st * 16 + c;
#pragma unroll
            for (int e = 0; e < 8; ++e) a[st][e]     = As[r][half * 16 + e];
#pragma unroll
            for (int e = 0; e < 8; ++e) a[st][8 + e] = As[r][half * 16 + 8 + e];
        }

#pragma unroll
        for (int t = 0; t < 4; ++t) {
            v16h bf;
#pragma unroll
            for (int e = 0; e < 16; ++e) bf[e] = Bs[t * 16 + c][half * 16 + e];
#pragma unroll
            for (int st = 0; st < 2; ++st)
                acc[st][t] = __builtin_amdgcn_wmma_f32_16x16x32_f16(
                    false, a[st], false, bf, (short)0, acc[st][t], false, false);
        }
    }

    // Epilogue: bias, scale, store f16 head-split.
    const int head = col0 >> 6;
#pragma unroll
    for (int st = 0; st < 2; ++st) {
#pragma unroll
        for (int t = 0; t < 4; ++t) {
#pragma unroll
            for (int j = 0; j < 8; ++j) {
                int gr = row0 + wv * 32 + st * 16 + j + 8 * half;
                int gc = col0 + t * 16 + c;
                int bb = gr >> 11, s = gr & (S_ - 1);     // S_ = 2048
                int hd = gc & 63;
                float v = (acc[st][t][j] + bias[gc]) * out_scale;
                size_t base = ((size_t)bb * NH_ + head);
                if (vt_layout)
                    out[(base * HD_ + hd) * S_ + s] = (_Float16)v;  // [B,NH,HD,S]
                else
                    out[(base * S_ + s) * HD_ + hd] = (_Float16)v;  // [B,NH,S,HD]
            }
        }
    }
}

// ---------------------------------------------------------------------------
// Kernel 2: flash attention.  One block per (b, h, 64-query strip).
// Q pre-scaled by 1/sqrt(HD).  V arrives pre-transposed [B,NH,HD,S], so both
// K and V^T tiles are staged with double-buffered GLOBAL_LOAD_ASYNC_TO_LDS.
// ---------------------------------------------------------------------------
__global__ __launch_bounds__(128)
void flash_attn_kernel(const _Float16* __restrict__ Q,   // [B,NH,S,HD]
                       const _Float16* __restrict__ K,   // [B,NH,S,HD]
                       const _Float16* __restrict__ V,   // [B,NH,HD,S]
                       const int* __restrict__ mask,     // [B,S,S]
                       _Float16* __restrict__ Wout)      // [B,S,NH,HD]
{
    __shared__ _Float16 Ks[2][64][64];   // [buf][key][hd]
    __shared__ _Float16 Vt[2][64][64];   // [buf][hd][key]
    __shared__ _Float16 Ps[64][64];      // [q_local][key_local], wave-private

    const int tid  = threadIdx.x;
    const int lane = tid & 31;
    const int wv   = tid >> 5;
    const int half = lane >> 4;
    const int c    = lane & 15;

    const int qb = blockIdx.x * 64;
    const int h  = blockIdx.y;
    const int b  = blockIdx.z;

    const _Float16* Qh  = Q + ((size_t)b * NH_ + h) * S_ * HD_;
    const _Float16* Kh  = K + ((size_t)b * NH_ + h) * S_ * HD_;
    const _Float16* VTh = V + ((size_t)b * NH_ + h) * HD_ * S_;

    // Async staging of one 64-key tile (K: [key][hd], V^T: [hd][key]).
    // 512 16B chunks per matrix; 128 threads -> 4 + 4 chunks per thread.
    auto stage = [&](int buf, int kb2) {
#pragma unroll
        for (int i = 0; i < 4; ++i) {
            int q = tid + i * 128;
            int row = q >> 3;             // key row (K) / hd row (V^T)
            int ch  = (q & 7) * 8;        // 8 f16 = 16B chunk
            async_b128(Kh  + (size_t)(kb2 + row) * HD_ + ch, &Ks[buf][row][ch]);
            async_b128(VTh + (size_t)row * S_ + kb2 + ch,    &Vt[buf][row][ch]);
        }
    };

    // Q A-fragments for this wave's 16 rows, two k-steps (hd 0..31, 32..63).
    const int qr = qb + wv * 16 + c;
    v16h qa[2];
#pragma unroll
    for (int ks = 0; ks < 2; ++ks) {
        const _Float16* base = Qh + (size_t)qr * HD_ + ks * 32;
#pragma unroll
        for (int e = 0; e < 8; ++e) qa[ks][e]     = base[half * 16 + e];
#pragma unroll
        for (int e = 0; e < 8; ++e) qa[ks][8 + e] = base[half * 16 + 8 + e];
    }

    float m[8], l[8];
    v8f o[4] = {};
#pragma unroll
    for (int j = 0; j < 8; ++j) { m[j] = -__builtin_inff(); l[j] = 0.f; }

    stage(0, 0);   // prologue: tile 0 into buffer 0

    for (int it = 0; it < S_ / 64; ++it) {
        const int kb  = it * 64;
        const int buf = it & 1;

        wait_async0();     // own copies of tile `it` landed
        __syncthreads();   // everyone's copies visible; buf^1 free to overwrite
        if (kb + 64 < S_) stage(buf ^ 1, kb + 64);

        // scores = Q . K^T : 4 column tiles x 2 k-steps
        v8f sc[4] = {};
#pragma unroll
        for (int ks = 0; ks < 2; ++ks) {
#pragma unroll
            for (int t = 0; t < 4; ++t) {
                v16h bf;
#pragma unroll
                for (int e = 0; e < 16; ++e)
                    bf[e] = Ks[buf][t * 16 + c][ks * 32 + half * 16 + e];
                sc[t] = __builtin_amdgcn_wmma_f32_16x16x32_f16(
                    false, qa[ks], false, bf, (short)0, sc[t], false, false);
            }
        }

        // mask + online softmax (row = j + 8*half, col = t*16 + c)
#pragma unroll
        for (int j = 0; j < 8; ++j) {
            const int grow = qb + wv * 16 + j + 8 * half;
            const int* mr = mask + ((size_t)b * S_ + grow) * S_ + kb;
            float s4[4];
            float tmax = -__builtin_inff();
#pragma unroll
            for (int t = 0; t < 4; ++t) {
                float sv = sc[t][j];
                if (mr[t * 16 + c] == 0) sv = -__builtin_inff();
                s4[t] = sv;
                tmax = fmaxf(tmax, sv);
            }
#pragma unroll
            for (int d = 1; d < 16; d <<= 1)
                tmax = fmaxf(tmax, __shfl_xor(tmax, d, 32));
            float mnew = fmaxf(m[j], tmax);
            float mcl  = fmaxf(mnew, -1e30f);          // guard all-masked rows
            float alpha = __expf(m[j] - mcl);
            float rsum = 0.f;
#pragma unroll
            for (int t = 0; t < 4; ++t) {
                float p = __expf(s4[t] - mcl);
                rsum += p;
                Ps[wv * 16 + j + 8 * half][t * 16 + c] = (_Float16)p;
            }
#pragma unroll
            for (int d = 1; d < 16; d <<= 1)
                rsum += __shfl_xor(rsum, d, 32);
            l[j] = l[j] * alpha + rsum;
            m[j] = mnew;
#pragma unroll
            for (int t = 0; t < 4; ++t) o[t][j] *= alpha;
        }

        // O += P . V  (P re-read from wave-private LDS in A layout; same-wave
        // DS ops are in order, no barrier needed)
#pragma unroll
        for (int ks = 0; ks < 2; ++ks) {
            v16h pa;
            const _Float16* pr = &Ps[wv * 16 + c][ks * 32];
#pragma unroll
            for (int e = 0; e < 8; ++e) pa[e]     = pr[half * 16 + e];
#pragma unroll
            for (int e = 0; e < 8; ++e) pa[8 + e] = pr[half * 16 + 8 + e];
#pragma unroll
            for (int t = 0; t < 4; ++t) {
                v16h bf;
#pragma unroll
                for (int e = 0; e < 16; ++e)
                    bf[e] = Vt[buf][t * 16 + c][ks * 32 + half * 16 + e];
                o[t] = __builtin_amdgcn_wmma_f32_16x16x32_f16(
                    false, pa, false, bf, (short)0, o[t], false, false);
            }
        }
    }

    // Finalize: normalize and store f16 into [B,S,NH,HD] (== [B,S,H]).
#pragma unroll
    for (int j = 0; j < 8; ++j) {
        float inv = (l[j] > 0.f) ? 1.0f / l[j] : 0.f;
        int grow = qb + wv * 16 + j + 8 * half;
#pragma unroll
        for (int t = 0; t < 4; ++t) {
            int hd = t * 16 + c;
            Wout[(((size_t)b * S_ + grow) * NH_ + h) * HD_ + hd] =
                (_Float16)(o[t][j] * inv);
        }
    }
}

// ---------------------------------------------------------------------------
// Kernel 3: output projection.  out(f32,[B*S,H]) = A(f16,[B*S,H]) * Wo + bo
// 128x64 tile; A tile staged with async global->LDS (pure f16 copy).
// ---------------------------------------------------------------------------
__global__ __launch_bounds__(128)
void out_gemm_kernel(const _Float16* __restrict__ A, const float* __restrict__ W,
                     const float* __restrict__ bias, float* __restrict__ out)
{
    __shared__ _Float16 As[128][32];  // 8 KB
    __shared__ _Float16 Bs[64][32];   // 4 KB

    const int tid  = threadIdx.x;
    const int lane = tid & 31;
    const int wv   = tid >> 5;
    const int half = lane >> 4;
    const int c    = lane & 15;

    const int row0 = blockIdx.x * 128;
    const int col0 = blockIdx.y * 64;

    const int    wkk  = tid >> 2;
    const int    wn0  = (tid & 3) * 16;
    const float* wsrc0 = &W[(size_t)wkk * H_ + col0 + wn0];

    v8f acc[2][4] = {};

    for (int k0 = 0; k0 < H_; k0 += 32) {
        __syncthreads();
        // --- A tile: 512 16B chunks via async copy; 4 chunks per thread
#pragma unroll
        for (int i = 0; i < 4; ++i) {
            int q = tid + i * 128;
            int row = q >> 2, ch = (q & 3) * 8;
            async_b128(A + (size_t)(row0 + row) * H_ + k0 + ch, &As[row][ch]);
        }
        // --- W tile: coalesced float4 loads, N-major scatter
        const float* wp = wsrc0 + (size_t)k0 * H_;
#pragma unroll
        for (int u = 0; u < 4; ++u) {
            v4f x = *(const v4f*)(wp + u * 4);
#pragma unroll
            for (int e2 = 0; e2 < 4; ++e2)
                Bs[wn0 + u * 4 + e2][wkk] = (_Float16)x[e2];
        }
        if (k0 + 32 < H_) __builtin_prefetch(wp + 32 * H_, 0, 1);
        wait_async0();
        __syncthreads();

        v16h a[2];
#pragma unroll
        for (int st = 0; st < 2; ++st) {
            const int r = wv * 32 + st * 16 + c;
#pragma unroll
            for (int e = 0; e < 8; ++e) a[st][e]     = As[r][half * 16 + e];
#pragma unroll
            for (int e = 0; e < 8; ++e) a[st][8 + e] = As[r][half * 16 + 8 + e];
        }

#pragma unroll
        for (int t = 0; t < 4; ++t) {
            v16h bf;
#pragma unroll
            for (int e = 0; e < 16; ++e) bf[e] = Bs[t * 16 + c][half * 16 + e];
#pragma unroll
            for (int st = 0; st < 2; ++st)
                acc[st][t] = __builtin_amdgcn_wmma_f32_16x16x32_f16(
                    false, a[st], false, bf, (short)0, acc[st][t], false, false);
        }
    }

#pragma unroll
    for (int st = 0; st < 2; ++st) {
#pragma unroll
        for (int t = 0; t < 4; ++t) {
#pragma unroll
            for (int j = 0; j < 8; ++j) {
                int gr = row0 + wv * 32 + st * 16 + j + 8 * half;
                int gc = col0 + t * 16 + c;
                out[(size_t)gr * H_ + gc] = acc[st][t][j] + bias[gc];
            }
        }
    }
}

// ---------------------------------------------------------------------------
// Launch: QKV projections -> flash attention -> output projection.
// Workspace: 4 x (B*S*H) f16 = 64 MB (Q, K head-split; V head-transposed;
// weighted).  Everything stays resident in the 192 MB L2.
// ---------------------------------------------------------------------------
extern "C" void kernel_launch(void* const* d_in, const int* in_sizes, int n_in,
                              void* d_out, int out_size, void* d_ws, size_t ws_size,
                              hipStream_t stream) {
    const float* query = (const float*)d_in[0];
    const float* key_  = (const float*)d_in[1];
    const float* value = (const float*)d_in[2];
    const int*   mask  = (const int*)d_in[3];
    const float* Wq = (const float*)d_in[4];
    const float* bq = (const float*)d_in[5];
    const float* Wk = (const float*)d_in[6];
    const float* bk = (const float*)d_in[7];
    const float* Wv = (const float*)d_in[8];
    const float* bv = (const float*)d_in[9];
    const float* Wo = (const float*)d_in[10];
    const float* bo = (const float*)d_in[11];
    float* out = (float*)d_out;

    const size_t elems = (size_t)B_ * S_ * H_;   // 8M
    _Float16* qws = (_Float16*)d_ws;
    _Float16* kws = qws + elems;
    _Float16* vws = kws + elems;
    _Float16* wws = vws + elems;

    dim3 blk(128);
    dim3 g1(B_ * S_ / 128, H_ / 64);
    qkv_gemm_kernel<<<g1, blk, 0, stream>>>(query, Wq, bq, qws, 0.125f, 0); // 1/sqrt(64)
    qkv_gemm_kernel<<<g1, blk, 0, stream>>>(key_,  Wk, bk, kws, 1.0f, 0);
    qkv_gemm_kernel<<<g1, blk, 0, stream>>>(value, Wv, bv, vws, 1.0f, 1);   // [B,NH,HD,S]

    dim3 g2(S_ / 64, NH_, B_);
    flash_attn_kernel<<<g2, blk, 0, stream>>>(qws, kws, vws, mask, wws);

    dim3 g3(B_ * S_ / 128, H_ / 64);
    out_gemm_kernel<<<g3, blk, 0, stream>>>(wws, Wo, bo, out);
}
